// Prot3DGraphModel_62294205661420
// MI455X (gfx1250) — matrix operand order repaired and checked by reading, not compiled, not run on hardware
//
#include <hip/hip_runtime.h>
#include <hip/hip_bf16.h>
#include <math.h>

typedef __attribute__((ext_vector_type(16))) _Float16 v16h;
typedef __attribute__((ext_vector_type(8)))  _Float16 v8h;
typedef __attribute__((ext_vector_type(8)))  float    v8f;

// ---------------------------------------------------------------------------
// Feature assembly: feat[N,1312] fp16 = [embed[seq] | node_s | seq_emb | 0pad]
// ---------------------------------------------------------------------------
__global__ void k_build_feat(const int* __restrict__ seq, const float* __restrict__ node_s,
                             const float* __restrict__ seq_emb, const float* __restrict__ embed,
                             _Float16* __restrict__ feat, int n) {
  long long idx = (long long)blockIdx.x * blockDim.x + threadIdx.x;
  long long total = (long long)n * 1312;
  if (idx >= total) return;
  int i = (int)(idx / 1312), j = (int)(idx % 1312);
  float v = 0.f;
  if (j < 20)        v = embed[seq[i] * 20 + j];
  else if (j < 26)   v = node_s[i * 6 + (j - 20)];
  else if (j < 1306) v = seq_emb[(long long)i * 1280 + (j - 26)];
  feat[idx] = (_Float16)v;
}

// dst[r,c<colsPad] = (c<cols) ? src[r,c] : 0   (fp32 -> fp16, column pad)
__global__ void k_pad_cols_f16(const float* __restrict__ src, _Float16* __restrict__ dst,
                               int rows, int cols, int colsPad) {
  long long idx = (long long)blockIdx.x * blockDim.x + threadIdx.x;
  long long total = (long long)rows * colsPad;
  if (idx >= total) return;
  int r = (int)(idx / colsPad), c = (int)(idx % colsPad);
  dst[idx] = (_Float16)((c < cols) ? src[(long long)r * cols + c] : 0.f);
}

// ---------------------------------------------------------------------------
// Weight swizzle: W[K,Cout] f32 -> fragment-tiled fp16 (K-rows zero-padded).
// Tile (kt, nt) = 32x16; within a tile, lane = (k%32)/16*16 + (c%16) owns 16
// contiguous halves indexed by k%16 — the contiguous k order (k = 16h + s) the
// GEMM's A loads also use, so the GEMM loads each fragment as 2x b128.
// ---------------------------------------------------------------------------
__global__ void k_wswz(const float* __restrict__ W, _Float16* __restrict__ dst,
                       int K, int Kpad, int Cout) {
  long long idx = (long long)blockIdx.x * blockDim.x + threadIdx.x;
  long long total = (long long)Kpad * Cout;
  if (idx >= total) return;
  int k = (int)(idx / Cout), c = (int)(idx % Cout);
  _Float16 v = (k < K) ? (_Float16)W[(long long)k * Cout + c] : (_Float16)0.f;
  int kt = k >> 5, kk = k & 31;
  int lane = ((kk >> 4) << 4) | (c & 15);
  long long o = (((long long)kt * (Cout >> 4) + (c >> 4)) * 32 + lane) * 16 + (kk & 15);
  dst[o] = v;
}

// ---------------------------------------------------------------------------
// WMMA GEMM: C[M,Nout] f32 = A[M,K] f16 @ Wsw (fragment-swizzled f16) (+bias)
// One wave per 32(M) x 64(N) tile: 2 A frags x 4 B frags -> 8 WMMAs / K-step.
// Requires: M%32==0, K%32==0, Nout%64==0 (guaranteed by padding).
// ---------------------------------------------------------------------------
__global__ void k_gemm_wmma(const _Float16* __restrict__ A, const _Float16* __restrict__ Wsw,
                            const float* __restrict__ bias, float* __restrict__ C,
                            int M, int K, int Nout) {
  const int lane  = threadIdx.x & 31;
  const int tileM = blockIdx.x << 5;
  const int tileN = blockIdx.y << 6;
  const int lid   = lane & 15;
  const int kA    = (lane < 16) ? 0 : 16;  // A: low lanes K 0..15, high 16..31
  const long long rowA0 = tileM + lid;
  const long long rowA1 = rowA0 + 16;
  const int ntiles = Nout >> 4;

  v8f acc00 = {}, acc01 = {}, acc02 = {}, acc03 = {};
  v8f acc10 = {}, acc11 = {}, acc12 = {}, acc13 = {};

  for (int k0 = 0; k0 < K; k0 += 32) {
    v16h a0, a1;
    {
      const _Float16* ap = A + rowA0 * K + k0 + kA;
      v8h lo = *(const v8h*)ap, hi = *(const v8h*)(ap + 8);
#pragma unroll
      for (int i = 0; i < 8; ++i) { a0[i] = lo[i]; a0[8 + i] = hi[i]; }
      ap = A + rowA1 * K + k0 + kA;
      lo = *(const v8h*)ap; hi = *(const v8h*)(ap + 8);
#pragma unroll
      for (int i = 0; i < 8; ++i) { a1[i] = lo[i]; a1[8 + i] = hi[i]; }
    }
    const _Float16* bp = Wsw + (((long long)(k0 >> 5) * ntiles + (tileN >> 4)) * 32 + lane) * 16;
    v16h b0, b1, b2, b3;
    {
      v8h lo, hi;
      lo = *(const v8h*)(bp +    0); hi = *(const v8h*)(bp +    8);
#pragma unroll
      for (int i = 0; i < 8; ++i) { b0[i] = lo[i]; b0[8 + i] = hi[i]; }
      lo = *(const v8h*)(bp +  512); hi = *(const v8h*)(bp +  520);
#pragma unroll
      for (int i = 0; i < 8; ++i) { b1[i] = lo[i]; b1[8 + i] = hi[i]; }
      lo = *(const v8h*)(bp + 1024); hi = *(const v8h*)(bp + 1032);
#pragma unroll
      for (int i = 0; i < 8; ++i) { b2[i] = lo[i]; b2[8 + i] = hi[i]; }
      lo = *(const v8h*)(bp + 1536); hi = *(const v8h*)(bp + 1544);
#pragma unroll
      for (int i = 0; i < 8; ++i) { b3[i] = lo[i]; b3[8 + i] = hi[i]; }
    }
    acc00 = __builtin_amdgcn_wmma_f32_16x16x32_f16(false, a0, false, b0, (short)0, acc00, false, false);
    acc10 = __builtin_amdgcn_wmma_f32_16x16x32_f16(false, a1, false, b0, (short)0, acc10, false, false);
    acc01 = __builtin_amdgcn_wmma_f32_16x16x32_f16(false, a0, false, b1, (short)0, acc01, false, false);
    acc11 = __builtin_amdgcn_wmma_f32_16x16x32_f16(false, a1, false, b1, (short)0, acc11, false, false);
    acc02 = __builtin_amdgcn_wmma_f32_16x16x32_f16(false, a0, false, b2, (short)0, acc02, false, false);
    acc12 = __builtin_amdgcn_wmma_f32_16x16x32_f16(false, a1, false, b2, (short)0, acc12, false, false);
    acc03 = __builtin_amdgcn_wmma_f32_16x16x32_f16(false, a0, false, b3, (short)0, acc03, false, false);
    acc13 = __builtin_amdgcn_wmma_f32_16x16x32_f16(false, a1, false, b3, (short)0, acc13, false, false);
  }

  float bb0 = 0.f, bb1 = 0.f, bb2 = 0.f, bb3 = 0.f;
  if (bias) {
    bb0 = bias[tileN + lid];      bb1 = bias[tileN + 16 + lid];
    bb2 = bias[tileN + 32 + lid]; bb3 = bias[tileN + 48 + lid];
  }
  const int rb = (lane < 16) ? 0 : 8;
#pragma unroll
  for (int v = 0; v < 8; ++v) {
    float* cr = C + (long long)(tileM + rb + v) * Nout + tileN + lid;
    cr[0] = acc00[v] + bb0; cr[16] = acc01[v] + bb1;
    cr[32] = acc02[v] + bb2; cr[48] = acc03[v] + bb3;
  }
#pragma unroll
  for (int v = 0; v < 8; ++v) {
    float* cr = C + (long long)(tileM + 16 + rb + v) * Nout + tileN + lid;
    cr[0] = acc10[v] + bb0; cr[16] = acc11[v] + bb1;
    cr[32] = acc12[v] + bb2; cr[48] = acc13[v] + bb3;
  }
}

// ---------------------------------------------------------------------------
// Attention (segment softmax) kernels
// ---------------------------------------------------------------------------
__device__ __forceinline__ unsigned int encF(float f) {
  unsigned int b = __float_as_uint(f);
  return (b & 0x80000000u) ? ~b : (b | 0x80000000u);   // monotonic uint key
}
__device__ __forceinline__ float decF(unsigned int u) {
  unsigned int b = (u & 0x80000000u) ? (u ^ 0x80000000u) : ~u;
  return __uint_as_float(b);
}

__global__ void k_edge_alpha(const float* __restrict__ q, const float* __restrict__ kk,
                             const float* __restrict__ ek, const int* __restrict__ src,
                             const int* __restrict__ dst, const unsigned char* __restrict__ em,
                             float* __restrict__ am, int E_, int d, float scale) {
  int e = blockIdx.x * (blockDim.x >> 5) + (threadIdx.x >> 5);
  int lane = threadIdx.x & 31;
  if (e >= E_) return;
  bool act = em ? (em[e] != 0) : true;
  float acc = 0.f;
  if (act) {
    long long qb = (long long)dst[e] * d, kb = (long long)src[e] * d, eb = (long long)e * d;
    for (int c = lane; c < d; c += 32)
      acc += q[qb + c] * (kk[kb + c] + ek[eb + c]);
#pragma unroll
    for (int o = 16; o; o >>= 1) acc += __shfl_xor(acc, o, 32);
  }
  if (lane == 0) am[e] = act ? acc * scale : -1e30f;
}

__global__ void k_segmax(const float* __restrict__ am, const int* __restrict__ dst,
                         unsigned int* __restrict__ mmax, int E_) {
  int e = blockIdx.x * blockDim.x + threadIdx.x;
  if (e >= E_) return;
  atomicMax(&mmax[dst[e]], encF(am[e]));
}

__global__ void k_fixm(const unsigned int* __restrict__ mmax, float* __restrict__ m, int n) {
  int i = blockIdx.x * blockDim.x + threadIdx.x;
  if (i >= n) return;
  float f = decF(mmax[i]);           // mmax==0 decodes to -NaN -> comparison false -> 0
  m[i] = (f > -1e30f) ? f : 0.f;
}

__global__ void k_edge_agg(const float* __restrict__ am, const float* __restrict__ m,
                           const float* __restrict__ vv, const float* __restrict__ ek,
                           const int* __restrict__ src, const int* __restrict__ dst,
                           const unsigned char* __restrict__ em,
                           float* __restrict__ denom, float* __restrict__ agg, int E_, int d) {
  int e = blockIdx.x * (blockDim.x >> 5) + (threadIdx.x >> 5);
  int lane = threadIdx.x & 31;
  if (e >= E_) return;
  bool act = em ? (em[e] != 0) : true;
  if (!act) return;
  int t = dst[e];
  float w = __expf(am[e] - m[t]);
  if (lane == 0) atomicAdd(&denom[t], w);
  long long vb = (long long)src[e] * d, eb = (long long)e * d, ab = (long long)t * d;
  for (int c = lane; c < d; c += 32)
    atomicAdd(&agg[ab + c], w * (vv[vb + c] + ek[eb + c]));
}

__global__ void k_finalize(const float* __restrict__ agg, const float* __restrict__ denom,
                           const float* __restrict__ skip, float* __restrict__ out,
                           int n, int d) {
  long long idx = (long long)blockIdx.x * blockDim.x + threadIdx.x;
  long long total = (long long)n * d;
  if (idx >= total) return;
  int nn = (int)(idx / d);
  float v = agg[idx] / (denom[nn] + 1e-16f) + skip[idx];
  out[idx] = (v > 0.f) ? v : 0.01f * v;   // leaky_relu, slope 0.01
}

// ---------------------------------------------------------------------------
// Masks + pooling
// ---------------------------------------------------------------------------
__global__ void k_epmask(const int* __restrict__ imp, unsigned char* __restrict__ ep, int n) {
  int i = blockIdx.x * blockDim.x + threadIdx.x;
  if (i < n) ep[i] = (imp[i] == 1) ? 1 : 0;
}
__global__ void k_edgemask(const unsigned char* __restrict__ ep, const int* __restrict__ src,
                           const int* __restrict__ dst, unsigned char* __restrict__ em, int E_) {
  int e = blockIdx.x * blockDim.x + threadIdx.x;
  if (e < E_) em[e] = ep[src[e]] & ep[dst[e]];
}
__global__ void k_count(const int* __restrict__ batch, float* __restrict__ cnt, int n) {
  int i = blockIdx.x * blockDim.x + threadIdx.x;
  if (i < n) atomicAdd(&cnt[batch[i]], 1.f);
}
__global__ void k_pool_accum(const float* __restrict__ h, const int* __restrict__ batch,
                             const unsigned char* __restrict__ ep, float* __restrict__ out,
                             int n, int d) {
  long long idx = (long long)blockIdx.x * blockDim.x + threadIdx.x;
  long long total = (long long)n * d;
  if (idx >= total) return;
  int nn = (int)(idx / d), c = (int)(idx % d);
  if (ep && !ep[nn]) return;               // x_epitope_padded: non-epitope rows are zero
  atomicAdd(&out[(long long)batch[nn] * d + c], h[idx]);
}
__global__ void k_pool_div(float* __restrict__ out, const float* __restrict__ cnt, int Bn, int d) {
  int idx = blockIdx.x * blockDim.x + threadIdx.x;
  if (idx >= 2 * Bn * d) return;
  int g = (idx / d) % Bn;
  out[idx] /= fmaxf(cnt[g], 1.f);
}

// ---------------------------------------------------------------------------
// Host orchestration
// ---------------------------------------------------------------------------
extern "C" void kernel_launch(void* const* d_in, const int* in_sizes, int n_in,
                              void* d_out, int out_size, void* d_ws, size_t ws_size,
                              hipStream_t stream) {
  (void)in_sizes; (void)n_in; (void)ws_size;
  constexpr int N = 20000, E = 320000, Bn = 16;

  const int*   seq        = (const int*)d_in[0];
  const float* node_s     = (const float*)d_in[1];
  const float* seq_emb    = (const float*)d_in[2];
  const int*   eidx       = (const int*)d_in[3];
  const int*   src        = eidx;
  const int*   dst        = eidx + E;
  const float* edge_s     = (const float*)d_in[4];
  const int*   importance = (const int*)d_in[5];
  const int*   batch      = (const int*)d_in[6];
  const float* embed      = (const float*)d_in[8];
  const float* pnW        = (const float*)d_in[9];
  const float* pnb        = (const float*)d_in[10];
  const float* peW        = (const float*)d_in[11];
  const float* peb        = (const float*)d_in[12];

  struct Layer { const float *Wq,*bq,*Wk,*bk,*Wv,*bv,*We,*Ws,*bs; };
  auto getLayer = [&](int b)->Layer {
    return Layer{ (const float*)d_in[b+0], (const float*)d_in[b+1], (const float*)d_in[b+2],
                  (const float*)d_in[b+3], (const float*)d_in[b+4], (const float*)d_in[b+5],
                  (const float*)d_in[b+6], (const float*)d_in[b+7], (const float*)d_in[b+8] };
  };
  Layer prot[3] = { getLayer(13), getLayer(22), getLayer(31) };
  Layer pock[3] = { getLayer(40), getLayer(49), getLayer(58) };

  // ---- workspace (bump allocator, 256B aligned, with overlays) ----
  char* wsb = (char*)d_ws;
  size_t off = 0;
  auto alloc = [&](size_t bytes)->void* {
    off = (off + 255) & ~(size_t)255;
    void* p = wsb + off; off += bytes; return p;
  };
  // ek region [E,256] f32; overlays: ea = first half, padded edge_s fp16 = second half
  float*    ekb   = (float*)alloc((size_t)E * 256 * 4);
  float*    ea    = ekb;
  _Float16* edgeH = (_Float16*)((char*)ekb + (size_t)E * 128 * 4);
  // q/k/v/skip region [4][N,256] f32; overlay: featH [N,1312] fp16 (dead before q written)
  float*    qreg  = (float*)alloc((size_t)4 * N * 256 * 4);
  float*    qb = qreg, *kb = qreg + (size_t)N*256, *vb = qreg + (size_t)2*N*256, *sk = qreg + (size_t)3*N*256;
  _Float16* featH = (_Float16*)qreg;
  _Float16* eaH   = (_Float16*)alloc((size_t)E * 128 * 2);
  _Float16* xH    = (_Float16*)alloc((size_t)N * 256 * 2);
  _Float16* wH    = (_Float16*)alloc((size_t)1312 * 256 * 2);
  float* x0    = (float*)alloc((size_t)N * 128 * 4);
  float* am    = (float*)alloc((size_t)E * 4);
  unsigned int* mmax = (unsigned int*)alloc((size_t)N * 4);
  float* mval  = (float*)alloc((size_t)N * 4);
  float* denom = (float*)alloc((size_t)N * 4);
  float* agg   = (float*)alloc((size_t)N * 256 * 4);
  float* hA    = (float*)alloc((size_t)N * 256 * 4);
  float* hB    = (float*)alloc((size_t)N * 256 * 4);
  unsigned char* ep = (unsigned char*)alloc(N);
  unsigned char* em = (unsigned char*)alloc(E);
  float* cnt   = (float*)alloc(Bn * 4);
  float* outp  = (float*)d_out;

  auto g1 = [](long long t){ return (unsigned)((t + 255) / 256); };

  // ---- edge projection: ea = pad64(edge_s) @ swz(peW) + peb ; then fp16 copy ----
  k_pad_cols_f16<<<g1((long long)E*64), 256, 0, stream>>>(edge_s, edgeH, E, 39, 64);
  k_wswz<<<g1((long long)64*128), 256, 0, stream>>>(peW, wH, 39, 64, 128);
  { dim3 g(E/32, 128/64); k_gemm_wmma<<<g, 32, 0, stream>>>(edgeH, wH, peb, ea, E, 64, 128); }
  k_pad_cols_f16<<<g1((long long)E*128), 256, 0, stream>>>(ea, eaH, E, 128, 128);

  // ---- input projection: x0 = feat @ swz(pnW) + pnb ----
  k_build_feat<<<g1((long long)N*1312), 256, 0, stream>>>(seq, node_s, seq_emb, embed, featH, N);
  k_wswz<<<g1((long long)1312*128), 256, 0, stream>>>(pnW, wH, 1306, 1312, 128);
  { dim3 g(N/32, 128/64); k_gemm_wmma<<<g, 32, 0, stream>>>(featH, wH, pnb, x0, N, 1312, 128); }

  // ---- masks + per-graph counts ----
  k_epmask<<<g1(N), 256, 0, stream>>>(importance, ep, N);
  k_edgemask<<<g1(E), 256, 0, stream>>>(ep, src, dst, em, E);
  hipMemsetAsync(cnt, 0, Bn * 4, stream);
  k_count<<<g1(N), 256, 0, stream>>>(batch, cnt, N);
  hipMemsetAsync(d_out, 0, (size_t)out_size * sizeof(float), stream);

  // ---- one TransformerConv layer ----
  auto runLayer = [&](const Layer& L, int cin, int cout, const float* xin, float* xout,
                      const unsigned char* mask) {
    k_pad_cols_f16<<<g1((long long)N*cin), 256, 0, stream>>>(xin, xH, N, cin, cin);
    dim3 gn(N/32, cout/64);
    long long tw = (long long)cin * cout;
    k_wswz<<<g1(tw), 256, 0, stream>>>(L.Wq, wH, cin, cin, cout);
    k_gemm_wmma<<<gn, 32, 0, stream>>>(xH, wH, L.bq, qb, N, cin, cout);
    k_wswz<<<g1(tw), 256, 0, stream>>>(L.Wk, wH, cin, cin, cout);
    k_gemm_wmma<<<gn, 32, 0, stream>>>(xH, wH, L.bk, kb, N, cin, cout);
    k_wswz<<<g1(tw), 256, 0, stream>>>(L.Wv, wH, cin, cin, cout);
    k_gemm_wmma<<<gn, 32, 0, stream>>>(xH, wH, L.bv, vb, N, cin, cout);
    k_wswz<<<g1(tw), 256, 0, stream>>>(L.Ws, wH, cin, cin, cout);
    k_gemm_wmma<<<gn, 32, 0, stream>>>(xH, wH, L.bs, sk, N, cin, cout);
    // ek = eaH @ We   (no bias)
    k_wswz<<<g1((long long)128*cout), 256, 0, stream>>>(L.We, wH, 128, 128, cout);
    dim3 ge(E/32, cout/64);
    k_gemm_wmma<<<ge, 32, 0, stream>>>(eaH, wH, nullptr, ekb, E, 128, cout);
    // segment softmax + aggregate
    float scale = 1.f / sqrtf((float)cout);
    int blkE = (E + 7) / 8;  // 8 waves (edges) per 256-thread block
    k_edge_alpha<<<blkE, 256, 0, stream>>>(qb, kb, ekb, src, dst, mask, am, E, cout, scale);
    hipMemsetAsync(mmax, 0, (size_t)N * 4, stream);
    hipMemsetAsync(denom, 0, (size_t)N * 4, stream);
    hipMemsetAsync(agg, 0, (size_t)N * cout * 4, stream);
    k_segmax<<<g1(E), 256, 0, stream>>>(am, dst, mmax, E);
    k_fixm<<<g1(N), 256, 0, stream>>>(mmax, mval, N);
    k_edge_agg<<<blkE, 256, 0, stream>>>(am, mval, vb, ekb, src, dst, mask, denom, agg, E, cout);
    k_finalize<<<g1((long long)N*cout), 256, 0, stream>>>(agg, denom, sk, xout, N, cout);
  };

  // ---- protein branch (full edge set), pool into out[0:4096] ----
  runLayer(prot[0], 128, 128, x0, hA, nullptr);
  runLayer(prot[1], 128, 256, hA, hB, nullptr);
  runLayer(prot[2], 256, 256, hB, hA, nullptr);
  k_pool_accum<<<g1((long long)N*256), 256, 0, stream>>>(hA, batch, nullptr, outp, N, 256);

  // ---- epitope branch (masked edges), pool masked rows into out[4096:8192] ----
  runLayer(pock[0], 128, 128, x0, hA, em);
  runLayer(pock[1], 128, 256, hA, hB, em);
  runLayer(pock[2], 256, 256, hB, hA, em);
  k_pool_accum<<<g1((long long)N*256), 256, 0, stream>>>(hA, batch, ep, outp + Bn*256, N, 256);

  k_pool_div<<<g1(2*Bn*256), 256, 0, stream>>>(outp, cnt, Bn, 256);
}
